// GNN_49366354100494
// MI455X (gfx1250) — matrix operand order repaired and checked by reading, not compiled, not run on hardware
//
#include <hip/hip_runtime.h>
#include <math.h>

typedef __bf16 bf16;
typedef __attribute__((ext_vector_type(16))) __bf16 v16bf;
typedef __attribute__((ext_vector_type(8)))  __bf16 v8bf;
typedef __attribute__((ext_vector_type(8)))  float  v8f;

#define NN 50000     // nodes  (divisible by 16)
#define NE 300000    // edges  (divisible by 16)
#define D  256
#define H  4
#define KG 2
#define LL 2

__device__ __forceinline__ float bf2f(bf16 b) { return (float)b; }
__device__ __forceinline__ bf16  f2bf(float f) { return (bf16)f; }
__device__ __forceinline__ float gelu_f(float x) {
  return 0.5f * x * (1.0f + erff(x * 0.70710678118654752f));
}

// ---- WMMA helpers (CDNA5 16x16x32 bf16, wave32) ----
// A: 16x32 (MxK) row-major (ld = D). Per lane: rows = lane&15, kh = lane>>4.
// a[0..7]  = A[row, kbase + kh*16 .. +7]       (16 contiguous bytes)
// a[8..15] = A[row, kbase + kh*16 + 8 .. +7]   (16 contiguous bytes)
__device__ __forceinline__ v16bf load_A16x32(const bf16* __restrict__ A,
                                             int row0, int kbase, int lane) {
  int row = row0 + (lane & 15);
  int kh = (lane >> 4) & 1;
  const bf16* p = A + (size_t)row * D + kbase + kh * 16;
  v8bf lo = *(const v8bf*)(p);
  v8bf hi = *(const v8bf*)(p + 8);
  return __builtin_shufflevector(lo, hi, 0, 1, 2, 3, 4, 5, 6, 7,
                                 8, 9, 10, 11, 12, 13, 14, 15);
}

// B: pre-swizzled packed layout. For tile (kt, ct): 32 lanes x 16 bf16 contiguous.
// element j of lane's vector = B[kt*32 + (lane>>4)*16 + j, ct*16 + (lane&15)]
__device__ __forceinline__ v16bf load_Bpk(const bf16* __restrict__ Bp, int kt, int ct,
                                          int lane) {
  return *(const v16bf*)(Bp + ((((size_t)kt * 16 + ct) * 32 + lane) << 4));
}

__device__ __forceinline__ v8f wmma_bf16(v16bf a, v16bf b, v8f c) {
  return __builtin_amdgcn_wmma_f32_16x16x32_bf16(false, a, false, b, (short)0, c,
                                                 false, false);
}

// ---------------- pack one [256,256] f32 weight into WMMA-ready bf16 ----------------
__global__ void pack_b_kernel(const float* __restrict__ src, bf16* __restrict__ dst) {
  int i = blockIdx.x * 256 + threadIdx.x;  // 65536 elements
  int j = i & 15;
  int lane = (i >> 4) & 31;
  int ct = (i >> 9) & 15;
  int kt = (i >> 13) & 7;
  int kh = lane >> 4;
  int col = ct * 16 + (lane & 15);
  int kr = kt * 32 + kh * 16 + j;
  dst[i] = f2bf(src[kr * D + col]);
}

// ---------------- elementwise / utility kernels ----------------
__global__ void cvt_f32_bf16_kernel(const float* __restrict__ s, bf16* __restrict__ d,
                                    long n) {
  long i = (long)blockIdx.x * blockDim.x + threadIdx.x;
  if (i < n) d[i] = f2bf(s[i]);
}

__global__ void fill_u32_kernel(unsigned* __restrict__ p, unsigned v, long n) {
  long i = (long)blockIdx.x * blockDim.x + threadIdx.x;
  if (i < n) p[i] = v;
}

// LayerNorm + cast to bf16: one block (256 thr) per node
__global__ void ln_kernel(const float* __restrict__ h, const float* __restrict__ g,
                          const float* __restrict__ b, bf16* __restrict__ out) {
  __shared__ float red[256];
  int n = blockIdx.x;
  int t = threadIdx.x;
  float x = h[(size_t)n * D + t];
  red[t] = x;
  __syncthreads();
  for (int s = 128; s > 0; s >>= 1) {
    if (t < s) red[t] += red[t + s];
    __syncthreads();
  }
  float mu = red[0] * (1.0f / D);
  __syncthreads();
  float dx = x - mu;
  red[t] = dx * dx;
  __syncthreads();
  for (int s = 128; s > 0; s >>= 1) {
    if (t < s) red[t] += red[t + s];
    __syncthreads();
  }
  float var = red[0] * (1.0f / D);
  float y = dx * rsqrtf(var + 1e-5f) * g[t] + b[t];
  out[(size_t)n * D + t] = f2bf(y);
}

// ---------------- node-level GEMM: C[M,256] = A[M,256] @ Bp (+bias) ----------------
// block = 256 thr (8 waves); wave w covers col tiles {2w, 2w+1}; grid = M/16
__global__ void node_gemm_kernel(const bf16* __restrict__ A, const bf16* __restrict__ Bp,
                                 const float* __restrict__ bias, bf16* __restrict__ C) {
  int lane = threadIdx.x & 31;
  int wave = threadIdx.x >> 5;
  int row0 = blockIdx.x * 16;
  int ct0 = wave * 2;
  v8f acc0 = {0, 0, 0, 0, 0, 0, 0, 0};
  v8f acc1 = {0, 0, 0, 0, 0, 0, 0, 0};
  for (int kt = 0; kt < 8; ++kt) {
    v16bf a = load_A16x32(A, row0, kt * 32, lane);
    if (kt < 7)
      __builtin_prefetch(A + (size_t)(row0 + (lane & 15)) * D + (kt + 1) * 32, 0, 0);
    acc0 = wmma_bf16(a, load_Bpk(Bp, kt, ct0, lane), acc0);
    acc1 = wmma_bf16(a, load_Bpk(Bp, kt, ct0 + 1, lane), acc1);
  }
  int kh = lane >> 4;
  int colA = ct0 * 16 + (lane & 15);
  float bvA = bias ? bias[colA] : 0.0f;
  float bvB = bias ? bias[colA + 16] : 0.0f;
#pragma unroll
  for (int i = 0; i < 8; ++i) {
    int r = row0 + i + 8 * kh;
    C[(size_t)r * D + colA] = f2bf(acc0[i] + bvA);
    C[(size_t)r * D + colA + 16] = f2bf(acc1[i] + bvB);
  }
}

// ---------------- edge transfer: xj_t = gelu(rel@WtR[kg] + NodeT[kg][src] + bt[kg]) ----------------
// block = 256 thr (8 waves); wave w covers col tiles {2w, 2w+1}, both KG branches
__global__ void edge_t_kernel(const bf16* __restrict__ relb, const bf16* __restrict__ W0p,
                              const bf16* __restrict__ W1p, const bf16* __restrict__ nodeT,
                              const float* __restrict__ bt, const int* __restrict__ src,
                              const int* __restrict__ kg_src, bf16* __restrict__ xj) {
  __shared__ int s_src[16], s_kg[16];
  int row0 = blockIdx.x * 16;
  if (threadIdx.x < 16) {
    int e = row0 + (int)threadIdx.x;
    s_src[threadIdx.x] = src[e];
    s_kg[threadIdx.x] = kg_src[e];
  }
  __syncthreads();
  int lane = threadIdx.x & 31;
  int wave = threadIdx.x >> 5;
  int ct0 = wave * 2;
  v8f c00 = {0, 0, 0, 0, 0, 0, 0, 0};  // branch0, tile0
  v8f c01 = {0, 0, 0, 0, 0, 0, 0, 0};  // branch0, tile1
  v8f c10 = {0, 0, 0, 0, 0, 0, 0, 0};  // branch1, tile0
  v8f c11 = {0, 0, 0, 0, 0, 0, 0, 0};  // branch1, tile1
  for (int kt = 0; kt < 8; ++kt) {
    v16bf a = load_A16x32(relb, row0, kt * 32, lane);
    c00 = wmma_bf16(a, load_Bpk(W0p, kt, ct0, lane), c00);
    c01 = wmma_bf16(a, load_Bpk(W0p, kt, ct0 + 1, lane), c01);
    c10 = wmma_bf16(a, load_Bpk(W1p, kt, ct0, lane), c10);
    c11 = wmma_bf16(a, load_Bpk(W1p, kt, ct0 + 1, lane), c11);
  }
  int kh = lane >> 4;
  int colA = ct0 * 16 + (lane & 15);
#pragma unroll
  for (int i = 0; i < 8; ++i) {
    int rl = i + 8 * kh;
    int e = row0 + rl;
    int kg = s_kg[rl];
    int sn = s_src[rl];
    const bf16* tp = nodeT + ((size_t)kg * NN + sn) * D;
    float v0 = (kg ? c10[i] : c00[i]) + bf2f(tp[colA]) + bt[kg * D + colA];
    float v1 = (kg ? c11[i] : c01[i]) + bf2f(tp[colA + 16]) + bt[kg * D + colA + 16];
    xj[(size_t)e * D + colA] = f2bf(gelu_f(v0));
    xj[(size_t)e * D + colA + 16] = f2bf(gelu_f(v1));
  }
}

// ---------------- fused k-projection + q.k attention logits ----------------
// block = 128 thr (4 waves), wave = head; 8 accumulators (4 col tiles x 2 branches)
__global__ void att_kernel(const bf16* __restrict__ xj, const bf16* __restrict__ Wk0p,
                           const bf16* __restrict__ Wk1p, const float* __restrict__ bk,
                           const bf16* __restrict__ nodeQ, const int* __restrict__ dst,
                           const int* __restrict__ kg_src, const int* __restrict__ kg_dst,
                           const float* __restrict__ beta, float* __restrict__ att) {
  __shared__ int s_dst[16], s_kgs[16], s_kgd[16];
  int row0 = blockIdx.x * 16;
  if (threadIdx.x < 16) {
    int e = row0 + (int)threadIdx.x;
    s_dst[threadIdx.x] = dst[e];
    s_kgs[threadIdx.x] = kg_src[e];
    s_kgd[threadIdx.x] = kg_dst[e];
  }
  __syncthreads();
  int lane = threadIdx.x & 31;
  int head = threadIdx.x >> 5;
  int kh = lane >> 4;
  v8f c0[4], c1[4];
#pragma unroll
  for (int t = 0; t < 4; ++t) {
    c0[t] = (v8f){0, 0, 0, 0, 0, 0, 0, 0};
    c1[t] = (v8f){0, 0, 0, 0, 0, 0, 0, 0};
  }
  for (int kt = 0; kt < 8; ++kt) {
    v16bf a = load_A16x32(xj, row0, kt * 32, lane);
#pragma unroll
    for (int t = 0; t < 4; ++t) {
      int ct = head * 4 + t;
      c0[t] = wmma_bf16(a, load_Bpk(Wk0p, kt, ct, lane), c0[t]);
      c1[t] = wmma_bf16(a, load_Bpk(Wk1p, kt, ct, lane), c1[t]);
    }
  }
  float part[8] = {0, 0, 0, 0, 0, 0, 0, 0};
#pragma unroll
  for (int t = 0; t < 4; ++t) {
    int col = (head * 4 + t) * 16 + (lane & 15);
#pragma unroll
    for (int i = 0; i < 8; ++i) {
      int rl = i + 8 * kh;
      int kgs = s_kgs[rl];
      float kv = (kgs ? c1[t][i] : c0[t][i]) + bk[kgs * D + col];
      float q = bf2f(nodeQ[((size_t)s_kgd[rl] * NN + s_dst[rl]) * D + col]);
      part[i] += q * kv;
    }
  }
  // reduce across the 16 lanes holding the 16 columns (stay within lane-halves)
#pragma unroll
  for (int m = 8; m >= 1; m >>= 1) {
#pragma unroll
    for (int i = 0; i < 8; ++i) part[i] += __shfl_xor(part[i], m, 32);
  }
  if ((lane & 15) == 0) {
#pragma unroll
    for (int i = 0; i < 8; ++i) {
      int e = row0 + i + 8 * kh;
      att[e * H + head] = part[i] * beta[e] * 0.125f;  // 1/sqrt(64)
    }
  }
}

// ---------------- scatter softmax (2 passes) ----------------
__device__ __forceinline__ unsigned f2key(float f) {
  unsigned u = __float_as_uint(f);
  return (u & 0x80000000u) ? ~u : (u | 0x80000000u);
}
__device__ __forceinline__ float key2f(unsigned k) {
  unsigned u = (k & 0x80000000u) ? (k ^ 0x80000000u) : ~k;
  return __uint_as_float(u);
}

__global__ void att_max_kernel(const float* __restrict__ att, const int* __restrict__ dst,
                               unsigned* __restrict__ mkey) {
  long i = (long)blockIdx.x * blockDim.x + threadIdx.x;
  if (i < (long)NE * H) {
    int e = (int)(i >> 2);
    int h = (int)(i & 3);
    atomicMax(&mkey[dst[e] * H + h], f2key(att[i]));
  }
}

__global__ void att_exp_kernel(float* __restrict__ att, const int* __restrict__ dst,
                               const unsigned* __restrict__ mkey,
                               float* __restrict__ denom) {
  long i = (long)blockIdx.x * blockDim.x + threadIdx.x;
  if (i < (long)NE * H) {
    int e = (int)(i >> 2);
    int h = (int)(i & 3);
    float m = key2f(mkey[dst[e] * H + h]);
    float ex = expf(att[i] - m);
    att[i] = ex;
    atomicAdd(&denom[dst[e] * H + h], ex);
  }
}

// ---------------- weighted scatter-sum of messages ----------------
__global__ void aggregate_kernel(const float* __restrict__ att,
                                 const float* __restrict__ denom,
                                 const bf16* __restrict__ xj, const int* __restrict__ dst,
                                 float* __restrict__ aggr) {
  long i = (long)blockIdx.x * blockDim.x + threadIdx.x;  // E * 64 items, 4 cols each
  if (i < (long)NE * 64) {
    int e = (int)(i >> 6);
    int c0 = ((int)(i & 63)) << 2;
    int h = c0 >> 6;
    int d = dst[e];
    float a = att[e * H + h] / (denom[d * H + h] + 1e-16f);
    const bf16* xp = xj + (size_t)e * D + c0;
    float* op = aggr + (size_t)d * D + c0;
#pragma unroll
    for (int j = 0; j < 4; ++j) atomicAdd(&op[j], a * bf2f(xp[j]));
  }
}

__global__ void final_kernel(const float* __restrict__ hin, const float* __restrict__ aggr,
                             float* __restrict__ hout) {
  long i = (long)blockIdx.x * blockDim.x + threadIdx.x;
  if (i < (long)NN * D) hout[i] = hin[i] + gelu_f(aggr[i]);
}

// ---------------- host orchestration ----------------
static inline int cdiv_l(long a, long b) { return (int)((a + b - 1) / b); }

extern "C" void kernel_launch(void* const* d_in, const int* in_sizes, int n_in,
                              void* d_out, int out_size, void* d_ws, size_t ws_size,
                              hipStream_t stream) {
  const float* x      = (const float*)d_in[0];
  const int*   ei     = (const int*)d_in[1];   // [2,E]
  const int*   ekg    = (const int*)d_in[2];   // [2,E]
  const float* beta   = (const float*)d_in[3]; // [E]
  const float* rel    = (const float*)d_in[4]; // [E,256]
  const float* ln_g   = (const float*)d_in[5]; // [L,256]
  const float* ln_b   = (const float*)d_in[6];
  const float* Wt     = (const float*)d_in[7];  // [L,K,512,256]
  const float* bt     = (const float*)d_in[8];  // [L,K,256]
  const float* Wq     = (const float*)d_in[9];  // [L,K,256,256]
  const float* bq     = (const float*)d_in[10];
  const float* Wk     = (const float*)d_in[11];
  const float* bk     = (const float*)d_in[12];

  const int* src = ei;
  const int* dst = ei + NE;
  const int* kg_src = ekg;
  const int* kg_dst = ekg + NE;

  // workspace carve-up (deterministic, ~546 MB)
  char* w = (char*)d_ws;
  auto carve = [&](size_t bytes) {
    char* p = w;
    w += (bytes + 255) & ~(size_t)255;
    return p;
  };
  const size_t MSZ = (size_t)D * D;  // 65536 elements per packed [256,256] matrix
  bf16*     hn    = (bf16*)carve((size_t)NN * D * 2);
  bf16*     relb  = (bf16*)carve((size_t)NE * D * 2);
  bf16*     wtaP  = (bf16*)carve(KG * MSZ * 2);  // Wt rows 0..255, packed
  bf16*     wtrP  = (bf16*)carve(KG * MSZ * 2);  // Wt rows 256..511, packed
  bf16*     wqP   = (bf16*)carve(KG * MSZ * 2);
  bf16*     wkP   = (bf16*)carve(KG * MSZ * 2);
  bf16*     nodeT = (bf16*)carve((size_t)KG * NN * D * 2);
  bf16*     nodeQ = (bf16*)carve((size_t)KG * NN * D * 2);
  bf16*     xj    = (bf16*)carve((size_t)NE * D * 2);
  float*    att   = (float*)carve((size_t)NE * H * 4);
  unsigned* mkey  = (unsigned*)carve((size_t)NN * H * 4);
  float*    denom = (float*)carve((size_t)NN * H * 4);
  float*    aggr  = (float*)carve((size_t)NN * D * 4);
  float*    hmid  = (float*)carve((size_t)NN * D * 4);
  (void)ws_size; (void)n_in; (void)in_sizes; (void)out_size;

  const unsigned NEG_MAX_KEY = 0x00800000u;  // encode(-FLT_MAX)
  const int PACK_BLOCKS = (int)(MSZ / 256);  // 256

  // relation embeddings to bf16 once (shared by both layers)
  {
    long n = (long)NE * D;
    cvt_f32_bf16_kernel<<<cdiv_l(n, 256), 256, 0, stream>>>(rel, relb, n);
  }

  for (int l = 0; l < LL; ++l) {
    const float* hin = (l == 0) ? x : hmid;
    float* hout = (l == 0) ? hmid : (float*)d_out;

    // per-layer weights -> packed WMMA-ready bf16 (8 [256,256] matrices)
    for (int k = 0; k < KG; ++k) {
      const float* wt_k = Wt + ((size_t)l * KG + k) * 512 * D;
      pack_b_kernel<<<PACK_BLOCKS, 256, 0, stream>>>(wt_k, wtaP + k * MSZ);
      pack_b_kernel<<<PACK_BLOCKS, 256, 0, stream>>>(wt_k + (size_t)256 * D,
                                                     wtrP + k * MSZ);
      pack_b_kernel<<<PACK_BLOCKS, 256, 0, stream>>>(
          Wq + ((size_t)l * KG + k) * D * D, wqP + k * MSZ);
      pack_b_kernel<<<PACK_BLOCKS, 256, 0, stream>>>(
          Wk + ((size_t)l * KG + k) * D * D, wkP + k * MSZ);
    }

    // layernorm -> hn (bf16)
    ln_kernel<<<NN, 256, 0, stream>>>(hin, ln_g + (size_t)l * D, ln_b + (size_t)l * D, hn);

    // node-level GEMMs: NodeT_k = hn @ Wt[k, :256, :],  NodeQ_k = hn @ Wq[k] + bq[k]
    for (int k = 0; k < KG; ++k) {
      node_gemm_kernel<<<NN / 16, 256, 0, stream>>>(hn, wtaP + k * MSZ, nullptr,
                                                    nodeT + (size_t)k * NN * D);
      node_gemm_kernel<<<NN / 16, 256, 0, stream>>>(hn, wqP + k * MSZ,
                                                    bq + (size_t)(l * KG + k) * D,
                                                    nodeQ + (size_t)k * NN * D);
    }

    // per-edge transfer (rel part of Wt, both branches + select) + gelu
    edge_t_kernel<<<NE / 16, 256, 0, stream>>>(relb, wtrP, wtrP + MSZ, nodeT,
                                               bt + (size_t)l * KG * D, src, kg_src, xj);

    // fused k-projection + q.k logits
    att_kernel<<<NE / 16, 128, 0, stream>>>(xj, wkP, wkP + MSZ,
                                            bk + (size_t)l * KG * D, nodeQ, dst, kg_src,
                                            kg_dst, beta, att);

    // reset softmax / aggregation state
    fill_u32_kernel<<<cdiv_l((long)NN * H, 256), 256, 0, stream>>>(mkey, NEG_MAX_KEY,
                                                                   (long)NN * H);
    fill_u32_kernel<<<cdiv_l((long)NN * H, 256), 256, 0, stream>>>((unsigned*)denom, 0u,
                                                                   (long)NN * H);
    fill_u32_kernel<<<cdiv_l((long)NN * D, 256), 256, 0, stream>>>((unsigned*)aggr, 0u,
                                                                   (long)NN * D);

    // scatter softmax
    att_max_kernel<<<cdiv_l((long)NE * H, 256), 256, 0, stream>>>(att, dst, mkey);
    att_exp_kernel<<<cdiv_l((long)NE * H, 256), 256, 0, stream>>>(att, dst, mkey, denom);

    // weighted scatter-sum of messages
    aggregate_kernel<<<cdiv_l((long)NE * 64, 256), 256, 0, stream>>>(att, denom, xj, dst,
                                                                     aggr);

    // residual + gelu
    final_kernel<<<cdiv_l((long)NN * D, 256), 256, 0, stream>>>(hin, aggr, hout);
  }
}
